// ModelKMeans_1623497638698
// MI455X (gfx1250) — matrix operand order repaired and checked by reading, not compiled, not run on hardware
//
#include <hip/hip_runtime.h>
#include <hip/hip_bf16.h>

// ---------------------------------------------------------------------------
// K-means on MI455X (gfx1250).
//   assign step  : S = X @ C^T via v_wmma_f32_16x16x32_bf16, argmin(||c||^2 - 2S)
//   B staging    : Tensor Data Mover (tensor_load_to_lds) double-buffered into
//                  LDS, shared by all 4 waves of the block (4x L2 traffic cut),
//                  synchronized with s_wait_tensorcnt + workgroup barriers.
//   update step  : fused into assign via global_atomic_add_f32 (single X pass)
// ---------------------------------------------------------------------------

#define DIMS 128
#define KC   512
#define NPTS 262144
#define ITERS 10
#define TILE_SHORTS 2048            // one 32x16-centroid bf16 B tile (4 chunks)
#define TILE_BYTES  4096

typedef __attribute__((ext_vector_type(16))) __bf16          v16bf;
typedef __attribute__((ext_vector_type(16))) unsigned short  v16u;
typedef __attribute__((ext_vector_type(8)))  float           v8f;
typedef __attribute__((ext_vector_type(4)))  unsigned int    u32x4;
typedef __attribute__((ext_vector_type(8)))  int             i32x8;
typedef __attribute__((ext_vector_type(4)))  int             i32x4;

#if __has_include(<hip/amd_detail/amd_gfx1250_TDM.h>)
#define TDM_6ARG 1
#endif

__device__ __forceinline__ unsigned short f32_to_bf16_rne(float f) {
  unsigned int u = __builtin_bit_cast(unsigned int, f);
  u += 0x7FFFu + ((u >> 16) & 1u);   // round-to-nearest-even
  return (unsigned short)(u >> 16);
}

// ---------------------------------------------------------------------------
// TDM: DMA one linear 4KB B tile from global to LDS. Descriptor per ISA
// CDNA5 §8.3/8.4: data_size=1B, tensor_dim0=tile_dim0=4096 (single row).
// Issued once per wave (EXEC ignored); tracked with TENSORcnt.
// ---------------------------------------------------------------------------
__device__ __forceinline__ void tdm_load_tile(const unsigned short* gsrc,
                                              unsigned short* lds_dst) {
  unsigned long long ga  = (unsigned long long)gsrc;          // 57-bit global
  unsigned int       lds = (unsigned int)(unsigned long long)lds_dst; // low 32 = LDS offset

  u32x4 g0 = {};
  g0[0] = 1u;                                          // count=1, no gather
  g0[1] = lds;                                         // lds_addr  [63:32]
  g0[2] = (unsigned int)(ga & 0xFFFFFFFFu);            // global_addr[95:64]
  g0[3] = (unsigned int)((ga >> 32) & 0x01FFFFFFu)     // global_addr[120:96]
        | 0x80000000u;                                 // type=2 [127:126]

  i32x8 g1 = {};
  g1[0] = 0;                                           // wg_mask=0, data_size=1B
  g1[1] = (int)((TILE_BYTES & 0xFFFF) << 16);          // tensor_dim0[15:0] @ [63:48]
  g1[2] = (int)((TILE_BYTES >> 16) & 0xFFFF);          // tensor_dim0[31:16] @ [79:64]
  g1[3] = (int)((TILE_BYTES & 0xFFFF) << 16);          // tile_dim0 @ [127:112]
  g1[4] = 0;                                           // tile_dim1/2 unused
  g1[5] = (int)TILE_BYTES;                             // tensor_dim0_stride[31:0]
  g1[6] = 0;
  g1[7] = 0;

  i32x4 z4 = {};
#ifdef TDM_6ARG
  i32x8 z8 = {};
  __builtin_amdgcn_tensor_load_to_lds(g0, g1, z4, z4, z8, 0);
#else
  __builtin_amdgcn_tensor_load_to_lds(g0, g1, z4, z4, 0);
#endif
}

// ---------------------------------------------------------------------------
// Pack fp32 centroids into bf16 B-tiles in the WMMA lane/VGPR swizzle.
// Packed index = ((ctile*4 + kchunk)*32 + lane)*16 + e.
// B 32x16 (KxN) 16-bit layout: lanes 0-15 hold K=0..7,16..23,
// lanes 16-31 hold K=8..15,24..31; N = lane % 16.
// ---------------------------------------------------------------------------
__global__ void pack_centroids(const float* __restrict__ C,
                               unsigned short* __restrict__ Cpk) {
  int tid = blockIdx.x * blockDim.x + threadIdx.x;     // [0, K*D)
  int e =  tid        & 15;
  int l = (tid >> 4)  & 31;
  int q = (tid >> 9)  & 3;
  int c =  tid >> 11;
  int n    = l & 15;
  int half = l >> 4;
  int k    = (e < 8) ? (half * 8 + e) : (16 + half * 8 + (e - 8)); // K index within chunk
  float v = C[(c * 16 + n) * DIMS + q * 32 + k];
  Cpk[tid] = f32_to_bf16_rne(v);
}

__global__ void centroid_norms(const float* __restrict__ C,
                               float* __restrict__ cnorm) {
  __shared__ float red[DIMS];
  int k = blockIdx.x, t = threadIdx.x;
  float v = C[k * DIMS + t];
  red[t] = v * v;
  __syncthreads();
  for (int s = DIMS / 2; s > 0; s >>= 1) {
    if (t < s) red[t] += red[t + s];
    __syncthreads();
  }
  if (t == 0) cnorm[k] = red[0];
}

__global__ void zero_buf(float* __restrict__ p, int n) {
  int i = blockIdx.x * blockDim.x + threadIdx.x;
  if (i < n) p[i] = 0.0f;
}

__global__ void update_centroids(float* __restrict__ C,
                                 const float* __restrict__ sums,
                                 const float* __restrict__ counts) {
  int i = blockIdx.x * blockDim.x + threadIdx.x;       // [0, K*D)
  C[i] = sums[i] / counts[i >> 7];                     // 0-count -> NaN (matches ref)
}

// ---------------------------------------------------------------------------
// Fused assign + accumulate. 128 threads = 4 waves; wave w owns rows
// [blockIdx.x*64 + w*16, +16). Uniform control flow around WMMA (EXEC all-1s).
// Wave 0 drives the TDM double-buffer of B tiles.
// ---------------------------------------------------------------------------
__global__ __launch_bounds__(128) void assign_update(
    const float* __restrict__ X,
    const unsigned short* __restrict__ Cpk,
    const float* __restrict__ cnorm,
    float* __restrict__ sums,
    float* __restrict__ counts,
    int* __restrict__ labels) {
  __shared__ float Xs[4][16][DIMS];                              // 32 KB
  __shared__ __align__(32) unsigned short Bs[2][TILE_SHORTS];    // 2 x 4 KB
  const int t = threadIdx.x;
  const int w = t >> 5;                                // wave 0..3
  const int l = t & 31;                                // lane
  const int rowbase = blockIdx.x * 64 + w * 16;

  // Kick off the TDM pipeline: tile 0 -> Bs[0] (wave 0 only, uniform branch).
  if (w == 0) tdm_load_tile(Cpk, &Bs[0][0]);

  // Stage this wave's 16x128 fp32 tile into LDS (coalesced float4).
  const float4* Xg = (const float4*)(X + (size_t)rowbase * DIMS);
  float4* Xl = (float4*)(&Xs[w][0][0]);
#pragma unroll
  for (int i = 0; i < 16; ++i) Xl[i * 32 + l] = Xg[i * 32 + l];
  __syncthreads();

  // Build 4 resident bf16 A chunks (16x32 each) in the ISA A-matrix swizzle:
  // lanes 0-15: M=0..15, V0..3 -> K=0..7, V4..7 -> K=16..23
  // lanes 16-31:           V0..3 -> K=8..15, V4..7 -> K=24..31
  const int m = l & 15, half = l >> 4;
  v16bf a[4];
#pragma unroll
  for (int q = 0; q < 4; ++q) {
    v16u uv;
#pragma unroll
    for (int e = 0; e < 16; ++e) {
      int vg = e >> 1, pos = e & 1;
      int k = (vg < 4) ? (half * 8 + vg * 2 + pos)
                       : (16 + half * 8 + (vg - 4) * 2 + pos);
      uv[e] = f32_to_bf16_rne(Xs[w][m][q * 32 + k]);
    }
    a[q] = __builtin_bit_cast(v16bf, uv);
  }

  float best[8];
  int   bidx[8];
#pragma unroll
  for (int i = 0; i < 8; ++i) { best[i] = 3.4e38f; bidx[i] = 0; }

  for (int c = 0; c < 32; ++c) {                       // 32 centroid tiles of 16
    const int p = c & 1;
    if (c + 1 < 32) {
      if (w == 0) {
        tdm_load_tile(Cpk + (c + 1) * TILE_SHORTS, &Bs[p ^ 1][0]);
        __builtin_amdgcn_s_wait_tensorcnt(1);          // tile c landed in LDS
      }
    } else {
      if (w == 0) __builtin_amdgcn_s_wait_tensorcnt(0);
    }
    __syncthreads();                                   // publish tile c to all waves

    v8f acc = {};
    const v16bf* Bl = (const v16bf*)&Bs[p][0];
#pragma unroll
    for (int q = 0; q < 4; ++q) {
      acc = __builtin_amdgcn_wmma_f32_16x16x32_bf16(
          false, a[q], false, Bl[q * 32 + l], (short)0, acc, false, false);
    }
    // acc[i]: row = (l<16 ? i : i+8), col = l%16. argmin over ||c||^2 - 2*S.
    float cn = cnorm[c * 16 + m];
#pragma unroll
    for (int i = 0; i < 8; ++i) {
      float d2 = cn - 2.0f * acc[i];
      if (d2 < best[i]) { best[i] = d2; bidx[i] = c * 16 + m; }
    }
    __syncthreads();                                   // done reading Bs[p]
  }

  // Cross-lane argmin over the 16 lanes sharing each row; lowest-index tie-break.
#pragma unroll
  for (int i = 0; i < 8; ++i) {
    float bv = best[i];
    int   bi = bidx[i];
#pragma unroll
    for (int off = 1; off < 16; off <<= 1) {
      float ov = __shfl_xor(bv, off, 32);
      int   oi = __shfl_xor(bi, off, 32);
      if (ov < bv || (ov == bv && oi < bi)) { bv = ov; bi = oi; }
    }
    best[i] = bv; bidx[i] = bi;
  }

  // Write labels + fused cluster accumulation.
#pragma unroll
  for (int i = 0; i < 8; ++i) {
    int labLo = __shfl(bidx[i], 0, 32);                // row i     (lanes 0-15)
    int labHi = __shfl(bidx[i], 16, 32);               // row i + 8 (lanes 16-31)
    if (l == 0)  { labels[rowbase + i]     = labLo; atomicAdd(&counts[labLo], 1.0f); }
    if (l == 16) { labels[rowbase + 8 + i] = labHi; atomicAdd(&counts[labHi], 1.0f); }
#pragma unroll
    for (int j = 0; j < 4; ++j) {
      atomicAdd(&sums[labLo * DIMS + l * 4 + j], Xs[w][i][l * 4 + j]);
      atomicAdd(&sums[labHi * DIMS + l * 4 + j], Xs[w][8 + i][l * 4 + j]);
    }
  }
}

extern "C" void kernel_launch(void* const* d_in, const int* in_sizes, int n_in,
                              void* d_out, int out_size, void* d_ws, size_t ws_size,
                              hipStream_t stream) {
  (void)in_sizes; (void)n_in; (void)out_size; (void)ws_size;
  const float* X = (const float*)d_in[0];
  int* labels = (int*)d_out;                           // int32 labels

  // Workspace layout (floats unless noted):
  float* C      = (float*)d_ws;                        // K*D       = 65536
  float* sums   = C + KC * DIMS;                       // K*D       = 65536
  float* counts = sums + KC * DIMS;                    // K         = 512
  float* cnorm  = counts + KC;                         // K         = 512
  unsigned short* Cpk = (unsigned short*)(cnorm + KC); // K*D bf16  (32B-aligned)

  // C0 = X[:K, :]
  hipMemcpyAsync(C, X, (size_t)KC * DIMS * sizeof(float),
                 hipMemcpyDeviceToDevice, stream);

  for (int it = 0; it < ITERS; ++it) {
    pack_centroids<<<KC * DIMS / 256, 256, 0, stream>>>(C, Cpk);
    centroid_norms<<<KC, DIMS, 0, stream>>>(C, cnorm);
    zero_buf<<<(KC * DIMS + KC + 255) / 256, 256, 0, stream>>>(sums, KC * DIMS + KC);
    assign_update<<<NPTS / 64, 128, 0, stream>>>(X, Cpk, cnorm, sums, counts, labels);
    update_centroids<<<KC * DIMS / 256, 256, 0, stream>>>(C, sums, counts);
  }
}
